// SinkhornAttention_1958505087429
// MI455X (gfx1250) — matrix operand order repaired and checked by reading, not compiled, not run on hardware
//
#include <hip/hip_runtime.h>
#include <hip/hip_bf16.h>

// ---------------------------------------------------------------------------
// Types for WMMA fragments (wave32, gfx1250)
// ---------------------------------------------------------------------------
typedef __attribute__((ext_vector_type(16))) __bf16        v16bf;
typedef __attribute__((ext_vector_type(8)))  float         v8f;
typedef __attribute__((ext_vector_type(4)))  unsigned int  uvec4;

union FragBF {
    v16bf          v;
    uvec4          q[2];
    unsigned short s[16];
};

__device__ __forceinline__ unsigned short f2bf(float f) {
    unsigned int u = __float_as_uint(f);
    unsigned int r = u + 0x7FFFu + ((u >> 16) & 1u);   // round-to-nearest-even
    return (unsigned short)(r >> 16);
}
__device__ __forceinline__ float bf2f(unsigned short h) {
    return __uint_as_float(((unsigned int)h) << 16);
}

// ---------------------------------------------------------------------------
// WMMA GEMM:  C = scale * (A[MxK] . B^T) (+bias), bf16 inputs, f32 accumulate
//   MODE_PROJ  : A row-major (K contig), B row-major N x K (K contig), 1 batch
//   MODE_ATTN  : batched (z = batch), same layouts as PROJ
//   MODE_STAGE4: A = p (256x256 per batch b=z&7), B = Vp rows (k*16+ti)*8+b,
//                C written at rows ((m*16+ti)*8+b) -> (T,B,E) layout directly
// ---------------------------------------------------------------------------
constexpr int BM = 128, BN = 128, BK = 32;
constexpr int LDA_S = BK + 8;    // 40 bf16  (80 B rows, 16B aligned)
constexpr int LDB_S = BN + 8;    // 136 bf16 (272 B rows, 16B aligned)

enum { MODE_PROJ = 0, MODE_ATTN = 1, MODE_STAGE4 = 2 };

template <int MODE, bool STORE_BF16, bool HAS_BIAS>
__global__ __launch_bounds__(256) void gemm_wmma(
    const unsigned short* __restrict__ A,
    const unsigned short* __restrict__ Bm,
    void* __restrict__ C,
    const float* __restrict__ bias,
    float scale, int M, int N, int K,
    long lda, long ldb, long ldc)
{
    __shared__ __align__(16) unsigned short sA[BM * LDA_S];
    __shared__ __align__(16) unsigned short sB[(MODE == MODE_STAGE4) ? (BK * LDB_S)
                                                                     : (BN * LDA_S)];

    const int tid   = threadIdx.x;
    const int lane  = tid & 31;
    const int wave  = tid >> 5;
    const int wm    = wave >> 2;       // 0..1 : 64 rows each
    const int wn    = wave & 3;        // 0..3 : 32 cols each
    const int lhalf = lane >> 4;       // lane half (K split)
    const int l16   = lane & 15;

    const int m0 = blockIdx.y * BM;
    const int n0 = blockIdx.x * BN;
    const int z  = blockIdx.z;

    const unsigned short* Ab = A;
    const unsigned short* Bb = Bm;
    char* Cb = (char*)C;
    if constexpr (MODE == MODE_ATTN) {
        Ab += (long)z * M * K;
        Bb += (long)z * N * K;
        Cb += (long)z * M * N * (STORE_BF16 ? 2 : 4);
    } else if constexpr (MODE == MODE_STAGE4) {
        const int b = z & 7, ti = z >> 3;
        Ab += (long)b * 256 * 256;                         // p[b]
        Bb += ((long)ti * 8 + b) * 1024;                   // Vp row (k*16+ti)*8+b
        Cb += (((long)ti * 8 + b) * 1024) * (STORE_BF16 ? 2 : 4);
    }

    v8f acc[4][2];
    const v8f vzero = {};
    #pragma unroll
    for (int i = 0; i < 4; i++)
        #pragma unroll
        for (int j = 0; j < 2; j++) acc[i][j] = vzero;

    const int mBase = wm * 64;
    const int nBase = wn * 32;

    for (int k0 = 0; k0 < K; k0 += BK) {
        __syncthreads();
        // ---- A tile: BM x BK (K contiguous) -> 512 x 16B chunks
        #pragma unroll
        for (int i = 0; i < 2; i++) {
            int sgm = tid + i * 256;
            int r = sgm >> 2, c = sgm & 3;
            *(uvec4*)&sA[r * LDA_S + c * 8] =
                *(const uvec4*)&Ab[(long)(m0 + r) * lda + k0 + c * 8];
        }
        // ---- B tile
        if constexpr (MODE == MODE_STAGE4) {
            // B is K x N (N contiguous): BK x BN
            #pragma unroll
            for (int i = 0; i < 2; i++) {
                int sgm = tid + i * 256;
                int r = sgm >> 4, c = sgm & 15;
                *(uvec4*)&sB[r * LDB_S + c * 8] =
                    *(const uvec4*)&Bb[(long)(k0 + r) * ldb + n0 + c * 8];
            }
        } else {
            // B is N x K (K contiguous): BN x BK
            #pragma unroll
            for (int i = 0; i < 2; i++) {
                int sgm = tid + i * 256;
                int r = sgm >> 2, c = sgm & 3;
                *(uvec4*)&sB[r * LDA_S + c * 8] =
                    *(const uvec4*)&Bb[(long)(n0 + r) * ldb + k0 + c * 8];
            }
        }
        __syncthreads();

        // ---- A fragments: lane l16 = M row; lhalf selects K 0..15 vs 16..31
        FragBF fa[4];
        #pragma unroll
        for (int tm = 0; tm < 4; tm++) {
            const unsigned short* p = &sA[(mBase + tm * 16 + l16) * LDA_S + lhalf * 16];
            fa[tm].q[0] = *(const uvec4*)p;          // K = base .. base+7
            fa[tm].q[1] = *(const uvec4*)(p + 8);    // K = base+8 .. base+15
        }
        // ---- B fragments: lane l16 = N col; K = lhalf*16 + j
        FragBF fb[2];
        if constexpr (MODE == MODE_STAGE4) {
            #pragma unroll
            for (int tn = 0; tn < 2; tn++) {
                const int n = nBase + tn * 16 + l16;
                #pragma unroll
                for (int j = 0; j < 16; j++)
                    fb[tn].s[j] = sB[(lhalf * 16 + j) * LDB_S + n];
            }
        } else {
            #pragma unroll
            for (int tn = 0; tn < 2; tn++) {
                const unsigned short* p = &sB[(nBase + tn * 16 + l16) * LDA_S + lhalf * 16];
                fb[tn].q[0] = *(const uvec4*)p;
                fb[tn].q[1] = *(const uvec4*)(p + 8);
            }
        }
        // ---- 8 WMMAs per wave per K-step
        #pragma unroll
        for (int tm = 0; tm < 4; tm++)
            #pragma unroll
            for (int tn = 0; tn < 2; tn++)
                acc[tm][tn] = __builtin_amdgcn_wmma_f32_16x16x32_bf16(
                    false, fa[tm].v, false, fb[tn].v, (short)0, acc[tm][tn],
                    false, false);
    }

    // ---- Epilogue: C/D layout m = r + 8*lhalf, n = l16
    #pragma unroll
    for (int tm = 0; tm < 4; tm++) {
        #pragma unroll
        for (int tn = 0; tn < 2; tn++) {
            #pragma unroll
            for (int r = 0; r < 8; r++) {
                const int gm = m0 + mBase + tm * 16 + r + 8 * lhalf;
                const int gn = n0 + nBase + tn * 16 + l16;
                float v = acc[tm][tn][r] * scale;
                if constexpr (HAS_BIAS) v += bias[gn];
                const long off = (long)gm * ldc + gn;
                if constexpr (STORE_BF16) ((unsigned short*)Cb)[off] = f2bf(v);
                else                      ((float*)Cb)[off] = v;
            }
        }
    }
}

// ---------------------------------------------------------------------------
// fp32 -> bf16 conversion (vectorized x4)
// ---------------------------------------------------------------------------
__global__ __launch_bounds__(256) void cvt_f32_bf16(const float* __restrict__ src,
                                                    unsigned short* __restrict__ dst,
                                                    long n)
{
    long i = ((long)blockIdx.x * 256 + threadIdx.x) * 4;
    if (i + 3 < n) {
        float4 f = *(const float4*)(src + i);
        dst[i + 0] = f2bf(f.x); dst[i + 1] = f2bf(f.y);
        dst[i + 2] = f2bf(f.z); dst[i + 3] = f2bf(f.w);
    }
}

// ---------------------------------------------------------------------------
// Bucket mean pool: dst[b][tb][e] = mean_ti src[(tb*16+ti)*8 + b][e]   (bf16)
// ---------------------------------------------------------------------------
__global__ __launch_bounds__(256) void bucket_mean(const unsigned short* __restrict__ src,
                                                   unsigned short* __restrict__ dst)
{
    long i = (long)blockIdx.x * 256 + threadIdx.x;      // 8*256*1024 total
    int e  = (int)(i & 1023);
    int tb = (int)((i >> 10) & 255);
    int b  = (int)(i >> 18);
    const unsigned short* p = src + (((long)tb * 16 * 8) + b) * 1024 + e;
    float s = 0.f;
    #pragma unroll
    for (int ti = 0; ti < 16; ti++) s += bf2f(p[(long)ti * 8 * 1024]);
    dst[i] = f2bf(s * 0.0625f);
}

// ---------------------------------------------------------------------------
// Bucket key-padding mask: padded iff all 16 positions padded
// ---------------------------------------------------------------------------
__global__ __launch_bounds__(256) void kpm_bucket(const unsigned char* __restrict__ mask,
                                                  unsigned char* __restrict__ out)
{
    int i = blockIdx.x * 256 + threadIdx.x;             // 8*256
    if (i >= 2048) return;
    int b = i >> 8, sb = i & 255;
    const unsigned char* p = mask + (long)b * 4096 + sb * 16;
    unsigned char all = 1;
    #pragma unroll
    for (int j = 0; j < 16; j++) all &= (p[j] ? 1 : 0);
    out[i] = all;
}

// ---------------------------------------------------------------------------
// Sinkhorn: one block per batch; whole 256x256 f32 matrix lives in LDS
// (256*257*4 = 257 KB -- uses CDNA5's 320 KB WGP LDS). 32 waves x 8 rows,
// wave32 shuffle reductions. Applies mask + 1/tau on load, writes exp() as bf16.
// ---------------------------------------------------------------------------
__global__ __launch_bounds__(1024) void sinkhorn_kernel(const float* __restrict__ attn,
                                                        const unsigned char* __restrict__ kpmb,
                                                        unsigned short* __restrict__ p_out)
{
    constexpr int NB = 256, LD = 257;
    __shared__ float la[NB * LD];

    const int b   = blockIdx.x;
    const int tid = threadIdx.x;
    const float* src = attn + (long)b * NB * NB;
    const unsigned char* km = kpmb + b * NB;

    for (int i = tid; i < NB * NB; i += 1024) {
        int t = i >> 8, s = i & 255;
        float v = src[i];
        const bool pt = km[t] != 0, ps = km[s] != 0;
        if ((ps && !pt) || (pt && !ps)) v = -3.402823466e38f;
        la[t * LD + s] = v * 1.33333333333333333f;      // / tau (0.75)
    }
    __syncthreads();

    const int lane = tid & 31, wave = tid >> 5;
    for (int it = 0; it < 8; it++) {
        // ---- logsumexp over axis -1 (rows)
        #pragma unroll 1
        for (int rr = 0; rr < 8; rr++) {
            const int t = wave * 8 + rr;
            float* row = &la[t * LD];
            float x[8], m = -__builtin_inff();
            #pragma unroll
            for (int j = 0; j < 8; j++) { x[j] = row[lane + j * 32]; m = fmaxf(m, x[j]); }
            #pragma unroll
            for (int o = 16; o > 0; o >>= 1) m = fmaxf(m, __shfl_xor(m, o, 32));
            float s = 0.f;
            #pragma unroll
            for (int j = 0; j < 8; j++) s += __expf(x[j] - m);
            #pragma unroll
            for (int o = 16; o > 0; o >>= 1) s += __shfl_xor(s, o, 32);
            const float lse = m + __logf(s);
            if (lse >= -3.0e38f)
                #pragma unroll
                for (int j = 0; j < 8; j++) row[lane + j * 32] = x[j] - lse;
        }
        __syncthreads();
        // ---- logsumexp over axis -2 (cols); +1 pad keeps banks conflict-free
        #pragma unroll 1
        for (int rr = 0; rr < 8; rr++) {
            const int c = wave * 8 + rr;
            float x[8], m = -__builtin_inff();
            #pragma unroll
            for (int j = 0; j < 8; j++) { x[j] = la[(lane + j * 32) * LD + c]; m = fmaxf(m, x[j]); }
            #pragma unroll
            for (int o = 16; o > 0; o >>= 1) m = fmaxf(m, __shfl_xor(m, o, 32));
            float s = 0.f;
            #pragma unroll
            for (int j = 0; j < 8; j++) s += __expf(x[j] - m);
            #pragma unroll
            for (int o = 16; o > 0; o >>= 1) s += __shfl_xor(s, o, 32);
            const float lse = m + __logf(s);
            if (lse >= -3.0e38f)
                #pragma unroll
                for (int j = 0; j < 8; j++) la[(lane + j * 32) * LD + c] = x[j] - lse;
        }
        __syncthreads();
    }

    for (int i = tid; i < NB * NB; i += 1024) {
        int t = i >> 8, s = i & 255;
        p_out[(long)b * NB * NB + i] = f2bf(__expf(la[t * LD + s]));
    }
}

// ---------------------------------------------------------------------------
// Host side
// ---------------------------------------------------------------------------
extern "C" void kernel_launch(void* const* d_in, const int* in_sizes, int n_in,
                              void* d_out, int out_size, void* d_ws, size_t ws_size,
                              hipStream_t stream)
{
    (void)in_sizes; (void)n_in; (void)out_size; (void)ws_size;

    const float*         query = (const float*)d_in[0];
    const float*         key_t = (const float*)d_in[1];
    const float*         value = (const float*)d_in[2];
    const unsigned char* kpm   = (const unsigned char*)d_in[3];
    const float* Wq = (const float*)d_in[4];
    const float* bq = (const float*)d_in[5];
    const float* Wk = (const float*)d_in[6];
    const float* bk = (const float*)d_in[7];
    const float* Wv = (const float*)d_in[8];
    const float* bv = (const float*)d_in[9];
    const float* Wo = (const float*)d_in[10];
    const float* bo = (const float*)d_in[11];

    constexpr long B = 8, E = 1024, T = 4096;
    constexpr long M  = T * B;      // 32768 rows in (T,B,E) flattening
    constexpr int  NB = 256;        // buckets

    char*  ws  = (char*)d_ws;
    size_t off = 0;
    auto alloc = [&](size_t bytes) -> char* {
        char* p = ws + off;
        off += (bytes + 255) & ~(size_t)255;
        return p;
    };

    unsigned short* qx  = (unsigned short*)alloc(M * E * 2);
    unsigned short* kx  = (unsigned short*)alloc(M * E * 2);
    unsigned short* vx  = (unsigned short*)alloc(M * E * 2);
    unsigned short* wqb = (unsigned short*)alloc(E * E * 2);
    unsigned short* wkb = (unsigned short*)alloc(E * E * 2);
    unsigned short* wvb = (unsigned short*)alloc(E * E * 2);
    unsigned short* wob = (unsigned short*)alloc(E * E * 2);
    unsigned short* Qp  = (unsigned short*)alloc(M * E * 2);
    unsigned short* Kp  = (unsigned short*)alloc(M * E * 2);
    unsigned short* Vp  = (unsigned short*)alloc(M * E * 2);
    unsigned short* qb  = (unsigned short*)alloc(B * NB * E * 2);
    unsigned short* kb  = (unsigned short*)alloc(B * NB * E * 2);
    float*          att = (float*)alloc(B * NB * NB * 4);
    unsigned short* pbf = (unsigned short*)alloc(B * NB * NB * 2);
    unsigned char*  kpb = (unsigned char*)alloc(B * NB);
    unsigned short* out2 = qx;   // qx is dead after the Q projection -> reuse

    // 1) fp32 -> bf16 conversions
    cvt_f32_bf16<<<(int)(M * E / 1024), 256, 0, stream>>>(query, qx, M * E);
    cvt_f32_bf16<<<(int)(M * E / 1024), 256, 0, stream>>>(key_t, kx, M * E);
    cvt_f32_bf16<<<(int)(M * E / 1024), 256, 0, stream>>>(value, vx, M * E);
    cvt_f32_bf16<<<(int)(E * E / 1024), 256, 0, stream>>>(Wq, wqb, E * E);
    cvt_f32_bf16<<<(int)(E * E / 1024), 256, 0, stream>>>(Wk, wkb, E * E);
    cvt_f32_bf16<<<(int)(E * E / 1024), 256, 0, stream>>>(Wv, wvb, E * E);
    cvt_f32_bf16<<<(int)(E * E / 1024), 256, 0, stream>>>(Wo, wob, E * E);

    // 2) Q/K/V projections (WMMA)
    dim3 gP(E / BN, M / BM, 1);
    gemm_wmma<MODE_PROJ, true, true><<<gP, 256, 0, stream>>>(
        qx, wqb, Qp, bq, 1.0f, (int)M, (int)E, (int)E, E, E, E);
    gemm_wmma<MODE_PROJ, true, true><<<gP, 256, 0, stream>>>(
        kx, wkb, Kp, bk, 1.0f, (int)M, (int)E, (int)E, E, E, E);
    gemm_wmma<MODE_PROJ, true, true><<<gP, 256, 0, stream>>>(
        vx, wvb, Vp, bv, 1.0f, (int)M, (int)E, (int)E, E, E, E);

    // 3) bucket aggregation + mask buckets
    bucket_mean<<<(int)(B * NB * E / 256), 256, 0, stream>>>(Qp, qb);
    bucket_mean<<<(int)(B * NB * E / 256), 256, 0, stream>>>(Kp, kb);
    kpm_bucket<<<8, 256, 0, stream>>>(kpm, kpb);

    // 4) attn = qb . kb^T * E^-0.5 (batched WMMA)
    dim3 gA(NB / BN, NB / BM, (unsigned)B);
    gemm_wmma<MODE_ATTN, false, false><<<gA, 256, 0, stream>>>(
        qb, kb, att, nullptr, 0.03125f, NB, NB, (int)E, E, E, NB);

    // 5) Sinkhorn normalization fully in LDS, emits p (bf16)
    sinkhorn_kernel<<<(unsigned)B, 1024, 0, stream>>>(att, kpb, pbf);

    // 6) out2[(t*B+b), e] = sum_sb p[b,tb,sb] * Vp[(sb*16+ti)*B+b, e]
    dim3 gS(E / BN, NB / BM, (unsigned)(B * 16));
    gemm_wmma<MODE_STAGE4, true, false><<<gS, 256, 0, stream>>>(
        pbf, Vp, out2, nullptr, 1.0f, NB, (int)E, NB, NB, 16 * B * E, 16 * B * E);

    // 7) final projection -> d_out (fp32, (T,B,E))
    dim3 gO(E / BN, M / BM, 1);
    gemm_wmma<MODE_PROJ, false, true><<<gO, 256, 0, stream>>>(
        out2, wob, (float*)d_out, bo, 1.0f, (int)M, (int)E, (int)E, E, E, E);
}
